// Encoder_57956288692354
// MI455X (gfx1250) — matrix operand order repaired and checked by reading, not compiled, not run on hardware
//
#include <hip/hip_runtime.h>
#include <hip/hip_bf16.h>

// ---------------- problem constants (match reference) ----------------
#define N_NODES 4096
#define N_EDGES 131072
#define F_INC   16
#define HIDC    256
#define LATC    64
#define KKEEP   2048      // ceil(0.5*N)
#define NWORDS  128       // N_NODES/32 bitset words per row
#define LRELU   0.2f
#define MBLK    128       // GEMM M rows per workgroup (8 waves x 16)
#define KSTEP   32        // GEMM K per WMMA step

typedef __bf16 bf16;
typedef bf16  v16bf __attribute__((ext_vector_type(16)));
typedef bf16  v8bf  __attribute__((ext_vector_type(8)));
typedef float v8f   __attribute__((ext_vector_type(8)));
typedef int   v4i   __attribute__((vector_size(16)));   // 128-bit payload type for async LDS builtin

#define CDIV(a,b) (((a)+(b)-1)/(b))

#if __has_builtin(__builtin_amdgcn_global_load_async_to_lds_b128)
#define HAVE_ASYNC_LDS 1
#endif

// async (or fallback sync) 16-byte global -> LDS copy
__device__ __forceinline__ void stage_cp16(const void* g, void* l) {
#ifdef HAVE_ASYNC_LDS
  __builtin_amdgcn_global_load_async_to_lds_b128(
      (__attribute__((address_space(1))) v4i*)g,
      (__attribute__((address_space(3))) v4i*)l, 0, 0);
#else
  *(v8bf*)l = *(const v8bf*)g;
#endif
}

template <int Ncnt>
__device__ __forceinline__ void stage_wait() {
#ifdef HAVE_ASYNC_LDS
#if __has_builtin(__builtin_amdgcn_s_wait_asynccnt)
  __builtin_amdgcn_s_wait_asynccnt(Ncnt);
#else
  asm volatile("s_wait_asynccnt %0" ::"i"(Ncnt) : "memory");
#endif
#endif
}

// ---------------- device helpers ----------------
__device__ __forceinline__ float wred_sum(float v) {
  #pragma unroll
  for (int o = 16; o; o >>= 1) v += __shfl_xor(v, o, 32);
  return v;
}
__device__ __forceinline__ float wred_max(float v) {
  #pragma unroll
  for (int o = 16; o; o >>= 1) v = fmaxf(v, __shfl_xor(v, o, 32));
  return v;
}
__device__ __forceinline__ int wred_or(int v) {
  #pragma unroll
  for (int o = 16; o; o >>= 1) v |= __shfl_xor(v, o, 32);
  return v;
}
// order-preserving float->uint encoding for atomicMax-based segment max
__device__ __forceinline__ unsigned fenc(float x) {
  unsigned u = __float_as_uint(x);
  return (u >> 31) ? ~u : (u | 0x80000000u);
}
__device__ __forceinline__ float fdec(unsigned u) {
  u = (u >> 31) ? (u & 0x7FFFFFFFu) : ~u;
  return __uint_as_float(u);
}

// ---------------- packing kernels (fp32 -> padded bf16) ----------------
__global__ void pack_xf_kernel(const float* __restrict__ Hin, const float* __restrict__ pos,
                               bf16* __restrict__ XF, int NN, int HIN, int KP) {
  int t = blockIdx.x * blockDim.x + threadIdx.x;
  if (t >= NN * KP) return;
  int n = t / KP, k = t - n * KP;
  float v = 0.f;
  if (k < HIN)          v = Hin[(size_t)n * HIN + k];
  else if (k < HIN + 3) v = pos[n * 3 + (k - HIN)];
  XF[t] = (bf16)v;
}

// pack W [fin,C] into WMMA B-fragment order, K padded to KP.
// (((kt*NT + nt)*32 + lane)*16 + e) holds W[kt*32 + (lane>>4)*16 + e][nt*16 + (lane&15)]
__global__ void pack_w_kernel(const float* __restrict__ W, bf16* __restrict__ WP,
                              int fin, int KP, int C) {
  int t = blockIdx.x * blockDim.x + threadIdx.x;
  if (t >= KP * C) return;
  int e    = t & 15;
  int lane = (t >> 4) & 31;
  int rest = t >> 9;
  int NT   = C >> 4;
  int nt   = rest % NT;
  int kt   = rest / NT;
  int k = (kt << 5) + ((lane >> 4) << 4) + e;
  int n = (nt << 4) + (lane & 15);
  float v = (k < fin) ? W[(size_t)k * C + n] : 0.f;
  WP[t] = (bf16)v;
}

// ---------------- WMMA GEMM with async-LDS double-buffered pipeline ----------------
// Y[NN,C] = XF[NN,KP](bf16) @ WP(bf16, pre-packed B fragments) + bias
// Workgroup: 256 threads = 8 waves; computes a MBLK x 16 output block.
// Per K-step: A panel (128x32 bf16, 8KB) + B fragment (1KB) streamed into LDS with
// GLOBAL_LOAD_ASYNC_TO_LDS_B128 (ASYNCcnt), double-buffered: s_wait_asynccnt 3 releases
// the previous stage while the next stage's 3 async loads/wave are still in flight.
__global__ __launch_bounds__(256)
void gemm_wmma_kernel(const bf16* __restrict__ XF, const bf16* __restrict__ WP,
                      const float* __restrict__ bias, float* __restrict__ Y,
                      int NN, int KP, int C) {
  __shared__ __attribute__((aligned(16))) bf16 As[2][MBLK * KSTEP]; // 2 x 8KB
  __shared__ __attribute__((aligned(16))) bf16 Bs[2][32 * 16];      // 2 x 1KB
  int tid  = threadIdx.x;
  int lane = tid & 31;
  int wav  = tid >> 5;                       // 0..7: wave's 16-row strip
  int NT   = C >> 4;
  int blk  = blockIdx.x;
  int mb   = blk / NT;
  int nt   = blk - mb * NT;
  int m0   = mb * MBLK;
  int nk   = KP >> 5;

  // staging geometry: A row chunk = 16B (8 bf16); 128 rows x 4 chunks = 512 = 2/thread.
  int ar   = tid >> 2;                       // rows 0..63 (+64 for second chunk)
  int ac   = (tid & 3) << 3;                 // col offset in bf16 (0,8,16,24)
  int bc   = (tid & 63) << 3;                // B chunk (redundant x4, identical data)

  auto issue = [&](int kt, int bufi) {
    const bf16* gA0 = XF + (size_t)(m0 + ar) * KP + (kt << 5) + ac;
    const bf16* gA1 = XF + (size_t)(m0 + 64 + ar) * KP + (kt << 5) + ac;
    const bf16* gB  = WP + ((size_t)kt * NT + nt) * 512 + bc;
    stage_cp16(gA0, &As[bufi][ar * KSTEP + ac]);
    stage_cp16(gA1, &As[bufi][(64 + ar) * KSTEP + ac]);
    stage_cp16(gB,  &Bs[bufi][bc]);
  };

  v8f acc = {0.f, 0.f, 0.f, 0.f, 0.f, 0.f, 0.f, 0.f};
  int r  = (wav << 4) + (lane & 15);         // A row within block for this lane
  int kb = (lane >> 4) << 4;                 // lanes 16-31: K offset +16

  issue(0, 0);
  for (int kt = 0; kt < nk; ++kt) {
    int bufi = kt & 1;
    if (kt + 1 < nk) { issue(kt + 1, bufi ^ 1); stage_wait<3>(); }
    else             { stage_wait<0>(); }
    __syncthreads();
    // WMMA fragments from LDS (ds_load_b128)
    v8bf alo = *(const v8bf*)&As[bufi][r * KSTEP + kb];
    v8bf ahi = *(const v8bf*)&As[bufi][r * KSTEP + kb + 8];
    v8bf blo = *(const v8bf*)&Bs[bufi][lane * 16];
    v8bf bhi = *(const v8bf*)&Bs[bufi][lane * 16 + 8];
    v16bf a, b;
    #pragma unroll
    for (int i = 0; i < 8; ++i) { a[i] = alo[i]; a[8 + i] = ahi[i];
                                  b[i] = blo[i]; b[8 + i] = bhi[i]; }
    acc = __builtin_amdgcn_wmma_f32_16x16x32_bf16(false, a, false, b, (short)0, acc,
                                                  false, false);
    __syncthreads();
  }
  int n = nt * 16 + (lane & 15);
  float bv = bias[n];
  int mrow = m0 + (wav << 4) + ((lane >> 4) << 3);  // C/D: VGPR r -> M = r (+8, lanes 16-31)
  #pragma unroll
  for (int rr = 0; rr < 8; ++rr)
    Y[(size_t)(mrow + rr) * C + n] = acc[rr] + bv;
}

// ---------------- sparse GATv2 edge kernels ----------------
__global__ void edge_score_kernel(const float* __restrict__ XL, const float* __restrict__ XR,
                                  const float* __restrict__ We, const float* __restrict__ be,
                                  const float* __restrict__ att,
                                  const int* __restrict__ src, const int* __restrict__ dst,
                                  const int* __restrict__ valid, const float* __restrict__ pos,
                                  int C, float* __restrict__ EW, unsigned* __restrict__ EMAXe) {
  int wid  = (int)((blockIdx.x * blockDim.x + threadIdx.x) >> 5);
  int lane = threadIdx.x & 31;
  if (wid >= N_EDGES) return;
  int s = src[wid], d = dst[wid];
  int vf = valid ? valid[wid] : 1;
  float e0 = pos[d * 3 + 0] - pos[s * 3 + 0];
  float e1 = pos[d * 3 + 1] - pos[s * 3 + 1];
  float e2 = pos[d * 3 + 2] - pos[s * 3 + 2];
  float acc = 0.f;
  for (int c = lane; c < C; c += 32) {
    float mv = XL[(size_t)s * C + c] + XR[(size_t)d * C + c]
             + e0 * We[c] + e1 * We[C + c] + e2 * We[2 * C + c] + be[c];
    float lr = mv > 0.f ? mv : LRELU * mv;
    acc += lr * att[c];
  }
  acc = wred_sum(acc);
  if (lane == 0) {
    float e = vf ? acc : -1e9f;
    EW[wid] = e;
    atomicMax(&EMAXe[d], fenc(e));
  }
}

__global__ void edge_exp_kernel(const int* __restrict__ dst, const int* __restrict__ valid,
                                const unsigned* __restrict__ EMAXe,
                                float* __restrict__ EW, float* __restrict__ EDEN) {
  int e = blockIdx.x * blockDim.x + threadIdx.x;
  if (e >= N_EDGES) return;
  int d = dst[e];
  float emax = fdec(EMAXe[d]);
  float vf = valid ? (float)valid[e] : 1.f;
  float ee = expf(EW[e] - emax) * vf;
  EW[e] = ee;
  atomicAdd(&EDEN[d], ee);
}

__global__ void edge_agg_kernel(const float* __restrict__ XL,
                                const int* __restrict__ src, const int* __restrict__ dst,
                                const float* __restrict__ EW, const float* __restrict__ EDEN,
                                float* __restrict__ OUT, int C) {
  int wid  = (int)((blockIdx.x * blockDim.x + threadIdx.x) >> 5);
  int lane = threadIdx.x & 31;
  if (wid >= N_EDGES) return;
  int s = src[wid], d = dst[wid];
  float alpha = EW[wid] / (EDEN[d] + 1e-16f);
  for (int c = lane; c < C; c += 32)
    atomicAdd(&OUT[(size_t)d * C + c], alpha * XL[(size_t)s * C + c]);
}

__global__ void finalize_kernel(const float* __restrict__ OUT, const float* __restrict__ bias,
                                float* __restrict__ H, int C, int total) {
  int t = blockIdx.x * blockDim.x + threadIdx.x;
  if (t >= total) return;
  float v = OUT[t] + bias[t % C];
  H[t] = v > 0.f ? v : (expf(v) - 1.f);
}

__global__ void fill_u32_kernel(unsigned* __restrict__ p, unsigned v, int n) {
  int i = blockIdx.x * blockDim.x + threadIdx.x;
  if (i < n) p[i] = v;
}

// ---------------- pooling: dense score over augmented adjacency ----------------
__global__ void pool_lin_kernel(const float* __restrict__ H, const float* __restrict__ pos,
                                const float* __restrict__ Wl, const float* __restrict__ bl,
                                const float* __restrict__ Wr, const float* __restrict__ br,
                                const float* __restrict__ We, const float* __restrict__ be,
                                float* __restrict__ SL, float* __restrict__ SR,
                                float* __restrict__ PE) {
  int nid  = (int)((blockIdx.x * blockDim.x + threadIdx.x) >> 5);
  int lane = threadIdx.x & 31;
  if (nid >= N_NODES) return;
  const int HIN = HIDC, fin = HIDC + 3;
  float al = 0.f, ar = 0.f;
  for (int k = lane; k < fin; k += 32) {
    float xv = (k < HIN) ? H[(size_t)nid * HIN + k] : pos[nid * 3 + (k - HIN)];
    al += xv * Wl[k];
    ar += xv * Wr[k];
  }
  al = wred_sum(al); ar = wred_sum(ar);
  if (lane == 0) {
    SL[nid] = al + bl[0];
    SR[nid] = ar + br[0];
    PE[nid] = pos[nid * 3 + 0] * We[0] + pos[nid * 3 + 1] * We[1]
            + pos[nid * 3 + 2] * We[2] + be[0];
  }
}

__global__ void adj_build_kernel(const int* __restrict__ src, const int* __restrict__ dst,
                                 unsigned* __restrict__ ADJ) {
  int e = blockIdx.x * blockDim.x + threadIdx.x;
  if (e >= N_EDGES) return;
  atomicOr(&ADJ[(size_t)src[e] * NWORDS + (dst[e] >> 5)], 1u << (dst[e] & 31));
}

// AUG[i] = ADJ[i] | OR_{k in nbr(i)} ADJ[k]   (bitset A + A@A, avoids 4096^3 matmul)
__global__ void adj_aug_kernel(const unsigned* __restrict__ ADJ, unsigned* __restrict__ AUG) {
  __shared__ unsigned rowi[NWORDS];
  int i = blockIdx.x, w = threadIdx.x;
  rowi[w] = ADJ[(size_t)i * NWORDS + w];
  __syncthreads();
  unsigned accum = rowi[w];
  for (int ww = 0; ww < NWORDS; ++ww) {
    unsigned bits = rowi[ww];
    while (bits) {
      int b = __ffs(bits) - 1;
      bits &= bits - 1;
      accum |= ADJ[(size_t)(ww * 32 + b) * NWORDS + w];
    }
  }
  AUG[(size_t)i * NWORDS + w] = accum;
}

__global__ void dense_score_kernel(const float* __restrict__ SL, const float* __restrict__ SR,
                                   const float* __restrict__ PE, const unsigned* __restrict__ AUG,
                                   const float* __restrict__ att, const float* __restrict__ bias,
                                   float* __restrict__ SCORE) {
  int j    = (int)((blockIdx.x * blockDim.x + threadIdx.x) >> 5);
  int lane = threadIdx.x & 31;
  if (j >= N_NODES) return;
  float srj = SR[j] + PE[j];
  float a0 = att[0];
  int wsel = j >> 5; unsigned bsel = 1u << (j & 31);
  float mx = -3e38f; int any = 0;
  for (int i = lane; i < N_NODES; i += 32) {
    int adj = (AUG[(size_t)i * NWORDS + wsel] & bsel) ? 1 : 0;
    float z = SL[i] + srj - PE[i];
    float e = (z > 0.f ? z : LRELU * z) * a0;
    e = adj ? e : -1e9f;
    any |= adj;
    mx = fmaxf(mx, e);
  }
  mx = wred_max(mx); any = wred_or(any);
  float s1 = 0.f, s2 = 0.f;
  for (int i = lane; i < N_NODES; i += 32) {
    int adj = (AUG[(size_t)i * NWORDS + wsel] & bsel) ? 1 : 0;
    float z = SL[i] + srj - PE[i];
    float e = (z > 0.f ? z : LRELU * z) * a0;
    e = adj ? e : -1e9f;
    float ex = expf(e - mx);
    s1 += ex; s2 += ex * SL[i];
  }
  s1 = wred_sum(s1); s2 = wred_sum(s2);
  if (lane == 0) {
    float out = any ? (s2 / s1) : 0.f;
    SCORE[j] = tanhf(out + bias[0]);
  }
}

// single-block bitonic sort of 4096 (score desc) in LDS -> PERM
__global__ void sort_topk_kernel(const float* __restrict__ SCORE, int* __restrict__ PERM) {
  __shared__ float key[N_NODES];
  __shared__ int   idx[N_NODES];
  int t = threadIdx.x;
  for (int i = t; i < N_NODES; i += 1024) { key[i] = SCORE[i]; idx[i] = i; }
  __syncthreads();
  for (int k = 2; k <= N_NODES; k <<= 1) {
    for (int j = k >> 1; j > 0; j >>= 1) {
      for (int i = t; i < N_NODES; i += 1024) {
        int ixj = i ^ j;
        if (ixj > i) {
          bool up = ((i & k) == 0);
          bool sw = up ? (key[i] < key[ixj]) : (key[i] > key[ixj]);  // descending overall
          if (sw) {
            float tk = key[i]; key[i] = key[ixj]; key[ixj] = tk;
            int ti = idx[i]; idx[i] = idx[ixj]; idx[ixj] = ti;
          }
        }
      }
      __syncthreads();
    }
  }
  for (int i = t; i < N_NODES; i += 1024) PERM[i] = idx[i];
}

__global__ void pool_gather_small_kernel(const int* __restrict__ PERM,
                                         const float* __restrict__ pos,
                                         float* __restrict__ POSP, int* __restrict__ INV) {
  int k = blockIdx.x * blockDim.x + threadIdx.x;
  if (k >= KKEEP) return;
  int p = PERM[k];
  INV[p] = k;
  POSP[k * 3 + 0] = pos[p * 3 + 0];
  POSP[k * 3 + 1] = pos[p * 3 + 1];
  POSP[k * 3 + 2] = pos[p * 3 + 2];
}

__global__ void pool_gather_h_kernel(const int* __restrict__ PERM, const float* __restrict__ SCORE,
                                     const float* __restrict__ H, float* __restrict__ HP) {
  int t = blockIdx.x * blockDim.x + threadIdx.x;
  if (t >= KKEEP * HIDC) return;
  int k = t / HIDC, c = t - k * HIDC;
  int p = PERM[k];
  HP[t] = H[(size_t)p * HIDC + c] * SCORE[p];
}

__global__ void remap_kernel(const int* __restrict__ src, const int* __restrict__ dst,
                             const int* __restrict__ INV, int* __restrict__ S2,
                             int* __restrict__ D2, int* __restrict__ V2) {
  int e = blockIdx.x * blockDim.x + threadIdx.x;
  if (e >= N_EDGES) return;
  int s = INV[src[e]], d = INV[dst[e]];
  int v = (s >= 0 && d >= 0) ? 1 : 0;
  S2[e] = v ? s : 0;
  D2[e] = v ? d : 0;
  V2[e] = v;
}

// ---------------- host-side layer driver ----------------
struct GatP { const float *Wl, *bl, *Wr, *br, *We, *be, *att, *bias; };
struct Scr  { bf16 *XF, *WPL, *WPR; float *XL, *XR, *OUT, *EW, *EDEN; unsigned* EMAXe; };

static void run_gat_layer(const float* Hin, int HIN, const float* posp, int NN,
                          const int* src, const int* dst, const int* valid,
                          const GatP& P, int C, float* Hout, const Scr& S,
                          unsigned encNeg, hipStream_t st) {
  int fin = HIN + 3;
  int KP  = (fin + 31) & ~31;
  int NT  = C >> 4;
  pack_xf_kernel<<<CDIV(NN * KP, 256), 256, 0, st>>>(Hin, posp, S.XF, NN, HIN, KP);
  pack_w_kernel<<<CDIV(KP * C, 256), 256, 0, st>>>(P.Wl, S.WPL, fin, KP, C);
  pack_w_kernel<<<CDIV(KP * C, 256), 256, 0, st>>>(P.Wr, S.WPR, fin, KP, C);
  int blocks = (NN / MBLK) * NT;
  gemm_wmma_kernel<<<blocks, 256, 0, st>>>(S.XF, S.WPL, P.bl, S.XL, NN, KP, C);
  gemm_wmma_kernel<<<blocks, 256, 0, st>>>(S.XF, S.WPR, P.br, S.XR, NN, KP, C);
  fill_u32_kernel<<<CDIV(NN, 256), 256, 0, st>>>(S.EMAXe, encNeg, NN);
  (void)hipMemsetAsync(S.EDEN, 0, (size_t)NN * 4, st);
  (void)hipMemsetAsync(S.OUT, 0, (size_t)NN * C * 4, st);
  edge_score_kernel<<<CDIV(N_EDGES * 32, 256), 256, 0, st>>>(S.XL, S.XR, P.We, P.be, P.att,
                                                             src, dst, valid, posp, C,
                                                             S.EW, S.EMAXe);
  edge_exp_kernel<<<CDIV(N_EDGES, 256), 256, 0, st>>>(dst, valid, S.EMAXe, S.EW, S.EDEN);
  edge_agg_kernel<<<CDIV(N_EDGES * 32, 256), 256, 0, st>>>(S.XL, src, dst, S.EW, S.EDEN,
                                                           S.OUT, C);
  finalize_kernel<<<CDIV(NN * C, 256), 256, 0, st>>>(S.OUT, P.bias, Hout, C, NN * C);
}

extern "C" void kernel_launch(void* const* d_in, const int* in_sizes, int n_in,
                              void* d_out, int out_size, void* d_ws, size_t ws_size,
                              hipStream_t stream) {
  (void)in_sizes; (void)n_in; (void)out_size; (void)ws_size;
  const float* x   = (const float*)d_in[0];
  const float* pos = (const float*)d_in[1];
  const int*   ei  = (const int*)d_in[2];
  const int* src = ei;
  const int* dst = ei + N_EDGES;
  auto gp = [&](int b) -> GatP {
    return GatP{ (const float*)d_in[b + 0], (const float*)d_in[b + 1],
                 (const float*)d_in[b + 2], (const float*)d_in[b + 3],
                 (const float*)d_in[b + 4], (const float*)d_in[b + 5],
                 (const float*)d_in[b + 6], (const float*)d_in[b + 7] };
  };
  GatP c0 = gp(3), c1 = gp(11), pg = gp(19), c2 = gp(27), c3 = gp(35);

  // ---- carve workspace (all offsets 256B aligned) ----
  char* wsp = (char*)d_ws; size_t off = 0;
  auto carve = [&](size_t bytes) -> void* {
    void* p = wsp + off; off = (off + bytes + 255) & ~(size_t)255; return p;
  };
  bf16*  XF    = (bf16*)carve((size_t)N_NODES * 288 * 2);
  bf16*  WPL   = (bf16*)carve((size_t)288 * 256 * 2);
  bf16*  WPR   = (bf16*)carve((size_t)288 * 256 * 2);
  float* XL    = (float*)carve((size_t)N_NODES * HIDC * 4);
  float* XR    = (float*)carve((size_t)N_NODES * HIDC * 4);
  float* OUT   = (float*)carve((size_t)N_NODES * HIDC * 4);
  float* HA    = (float*)carve((size_t)N_NODES * HIDC * 4);
  float* HB    = (float*)carve((size_t)N_NODES * HIDC * 4);
  float* HP    = (float*)carve((size_t)KKEEP * HIDC * 4);
  float* POSP  = (float*)carve((size_t)KKEEP * 3 * 4);
  float* EW    = (float*)carve((size_t)N_EDGES * 4);
  unsigned* EMAXe = (unsigned*)carve((size_t)N_NODES * 4);
  float* EDEN  = (float*)carve((size_t)N_NODES * 4);
  float* SL    = (float*)carve((size_t)N_NODES * 4);
  float* SR    = (float*)carve((size_t)N_NODES * 4);
  float* PEb   = (float*)carve((size_t)N_NODES * 4);
  float* SCORE = (float*)carve((size_t)N_NODES * 4);
  int*   PERM  = (int*)carve((size_t)N_NODES * 4);
  int*   INV   = (int*)carve((size_t)N_NODES * 4);
  int*   S2    = (int*)carve((size_t)N_EDGES * 4);
  int*   D2    = (int*)carve((size_t)N_EDGES * 4);
  int*   V2    = (int*)carve((size_t)N_EDGES * 4);
  unsigned* ADJ = (unsigned*)carve((size_t)N_NODES * NWORDS * 4);
  unsigned* AUG = (unsigned*)carve((size_t)N_NODES * NWORDS * 4);

  union { float f; unsigned u; } cv; cv.f = -1e9f;
  unsigned encNeg = (cv.u & 0x80000000u) ? ~cv.u : (cv.u | 0x80000000u);

  Scr S{XF, WPL, WPR, XL, XR, OUT, EW, EDEN, EMAXe};

  // conv0, conv1 on full graph
  run_gat_layer(x,  F_INC, pos, N_NODES, src, dst, nullptr, c0, HIDC, HA, S, encNeg, stream);
  run_gat_layer(HA, HIDC,  pos, N_NODES, src, dst, nullptr, c1, HIDC, HB, S, encNeg, stream);

  // SAGPool: dense score over (A + A^2) > 0 via bitsets
  pool_lin_kernel<<<CDIV(N_NODES * 32, 256), 256, 0, stream>>>(HB, pos, pg.Wl, pg.bl, pg.Wr,
                                                               pg.br, pg.We, pg.be, SL, SR, PEb);
  (void)hipMemsetAsync(ADJ, 0, (size_t)N_NODES * NWORDS * 4, stream);
  adj_build_kernel<<<CDIV(N_EDGES, 256), 256, 0, stream>>>(src, dst, ADJ);
  adj_aug_kernel<<<N_NODES, NWORDS, 0, stream>>>(ADJ, AUG);
  dense_score_kernel<<<CDIV(N_NODES * 32, 256), 256, 0, stream>>>(SL, SR, PEb, AUG, pg.att,
                                                                  pg.bias, SCORE);
  sort_topk_kernel<<<1, 1024, 0, stream>>>(SCORE, PERM);
  (void)hipMemsetAsync(INV, 0xFF, (size_t)N_NODES * 4, stream);   // INV = -1
  pool_gather_small_kernel<<<CDIV(KKEEP, 256), 256, 0, stream>>>(PERM, pos, POSP, INV);
  pool_gather_h_kernel<<<CDIV(KKEEP * HIDC, 256), 256, 0, stream>>>(PERM, SCORE, HB, HP);
  remap_kernel<<<CDIV(N_EDGES, 256), 256, 0, stream>>>(src, dst, INV, S2, D2, V2);

  // conv2, conv3 on pooled graph (valid-masked edges)
  run_gat_layer(HP, HIDC, POSP, KKEEP, S2, D2, V2, c2, HIDC, HA, S, encNeg, stream);
  run_gat_layer(HA, HIDC, POSP, KKEEP, S2, D2, V2, c3, LATC, (float*)d_out, S, encNeg, stream);
}
